// MolecularPredictionNetwork_86079734546997
// MI455X (gfx1250) — matrix operand order repaired and checked by reading, not compiled, not run on hardware
//
#include <hip/hip_runtime.h>

// ---------------------------------------------------------------------------
// Fused 3-layer MLP (300->300->300->1, ReLU) for MI455X / gfx1250.
//  - Layers 1&2 on v_wmma_f32_16x16x32_f16, fp32 accumulation.
//  - fp32 accuracy recovered via f16 hi/lo split: A*B ~= AhiBhi+AhiBlo+AloBhi
//    (dropped AloBlo term is ~2^-22 relative) -> 3 WMMAs per fp32 tile.
//  - X streams once from HBM (629 MB ~ 27us @ 23.3 TB/s); weights converted
//    once into d_ws and stay L2-resident; activations live in LDS only.
// ---------------------------------------------------------------------------

#define D_IN    300
#define D_HID   300
#define KPAD    320            // K padded to 10 WMMA k-steps of 32
#define NPAD    304            // N padded to 19 tiles of 16
#define NKT     10
#define NNT     19
#define TM      64             // atoms per block
#define WAVES   4
#define THREADS 128
#define PER     (NPAD * KPAD)  // elements per weight plane (97280)

typedef __attribute__((ext_vector_type(16))) _Float16 v16h;
typedef __attribute__((ext_vector_type(8)))  _Float16 v8h;
typedef __attribute__((ext_vector_type(8)))  float    v8f;
typedef __attribute__((ext_vector_type(4)))  float    v4f;
typedef __attribute__((ext_vector_type(4)))  unsigned int v4u;

union V16U { v16h v; v8h h[2]; };

// ---------------------------------------------------------------------------
// Prep: W[k][n] fp32 -> transposed, padded f16 hi/lo planes in d_ws:
//   [W1hi | W1lo | W2hi | W2lo], each [NPAD][KPAD], plane[n*KPAD + k].
// A B-fragment lane (column n, 16 consecutive k) is then 2 contiguous b128s.
// ---------------------------------------------------------------------------
__global__ void prep_weights_f16split(const float* __restrict__ W1,
                                      const float* __restrict__ W2,
                                      _Float16* __restrict__ wh) {
  int idx = blockIdx.x * blockDim.x + threadIdx.x;
  if (idx >= 2 * PER) return;
  int layer = idx / PER;
  int rem   = idx - layer * PER;
  int n = rem / KPAD;
  int k = rem - n * KPAD;
  const float* W = layer ? W2 : W1;
  float w = (n < D_HID && k < D_IN) ? W[k * D_HID + n] : 0.0f;
  _Float16 hi = (_Float16)w;
  _Float16 lo = (_Float16)(w - (float)hi);
  _Float16* base = wh + (size_t)layer * (2 * PER);
  base[rem]       = hi;
  base[PER + rem] = lo;
}

// ---------------------------------------------------------------------------
// One GEMM layer: D = relu(A @ B + bias), A (hi,lo f16) from LDS,
// B (hi,lo f16) from L2-resident global planes. Wave w owns n-tiles w, w+4,...
// Fragment layouts (16-bit A 16x32; B lane=column n): contiguous k = 16*(lane/16)+s in A and B.
// ---------------------------------------------------------------------------
template <bool STORE_F32>
__device__ __forceinline__ void gemm_layer(
    const _Float16* __restrict__ Ah, const _Float16* __restrict__ Al,
    const _Float16* __restrict__ Bh, const _Float16* __restrict__ Bl,
    const float* __restrict__ bias,
    _Float16* __restrict__ Dhi, _Float16* __restrict__ Dlo,
    float* __restrict__ Df)
{
  const int lane  = threadIdx.x & 31;
  const int wave  = threadIdx.x >> 5;
  const int lmod  = lane & 15;
  const int lhalf = lane >> 4;

  for (int nt = wave; nt < NNT; nt += WAVES) {      // wave-uniform: EXEC all 1s
    const int n  = nt * 16 + lmod;                  // n < NPAD always
    const int nc = n < D_HID ? n : (D_HID - 1);
    const float bn = (n < D_HID) ? bias[nc] : 0.0f; // zero bias in pad columns

    v8f acc[4] = {};                                // 4 m-tiles of 16 rows

#pragma unroll
    for (int ks = 0; ks < NKT; ++ks) {
      // B fragment: lane = column n; rows k0..k0+15 contiguous (32 B).
      const int kkB = ks * 32 + lhalf * 16;
      const v16h bh = *(const v16h*)(Bh + (size_t)n * KPAD + kkB);
      const v16h bl = *(const v16h*)(Bl + (size_t)n * KPAD + kkB);
      // A fragment: lane = row m; [0..15]=k0..k0+15 contiguous (matches B).
      const int kkA = ks * 32 + lhalf * 16;
#pragma unroll
      for (int mt = 0; mt < 4; ++mt) {
        const int m = mt * 16 + lmod;
        const _Float16* pah = Ah + m * KPAD + kkA;
        const _Float16* pal = Al + m * KPAD + kkA;
        V16U ah, al;
        ah.h[0] = *(const v8h*)pah;
        ah.h[1] = *(const v8h*)(pah + 8);
        al.h[0] = *(const v8h*)pal;
        al.h[1] = *(const v8h*)(pal + 8);
        acc[mt] = __builtin_amdgcn_wmma_f32_16x16x32_f16(
            false, ah.v, false, bh, (short)0, acc[mt], false, false);
        acc[mt] = __builtin_amdgcn_wmma_f32_16x16x32_f16(
            false, ah.v, false, bl, (short)0, acc[mt], false, false);
        acc[mt] = __builtin_amdgcn_wmma_f32_16x16x32_f16(
            false, al.v, false, bh, (short)0, acc[mt], false, false);
      }
    }

    // Epilogue: bias + ReLU; C/D layout: elem r of lane l -> row r+8*(l/16).
#pragma unroll
    for (int mt = 0; mt < 4; ++mt) {
#pragma unroll
      for (int r = 0; r < 8; ++r) {
        float v = acc[mt][r] + bn;
        v = v > 0.0f ? v : 0.0f;
        const int m = mt * 16 + r + lhalf * 8;
        if (STORE_F32) {
          Df[m * NPAD + n] = v;
        } else {
          _Float16 h = (_Float16)v;
          Dhi[m * KPAD + n] = h;
          Dlo[m * KPAD + n] = (_Float16)(v - (float)h);
        }
      }
    }
  }
}

__global__ __launch_bounds__(THREADS)
void mlp3_fused_wmma(const float* __restrict__ X,
                     const _Float16* __restrict__ Wh,
                     const float* __restrict__ b1,
                     const float* __restrict__ b2,
                     const float* __restrict__ W3,
                     const float* __restrict__ b3,
                     float* __restrict__ out,
                     int natoms)
{
  extern __shared__ unsigned char smem[];           // 160 KB dynamic LDS
  _Float16* Xhi = (_Float16*)smem;                  // [TM][KPAD]
  _Float16* Xlo = Xhi + TM * KPAD;
  _Float16* Hhi = Xlo + TM * KPAD;                  // layer-1 output (hi,lo)
  _Float16* Hlo = Hhi + TM * KPAD;
  float*    Hf  = (float*)smem;                     // layer-2 out, overlays X

  const int  tid  = threadIdx.x;
  const long row0 = (long)blockIdx.x * TM;

  // Zero all LDS so k-pad columns (300..319) are exact zeros.
  {
    v4u z = {0u, 0u, 0u, 0u};
    v4u* p = (v4u*)smem;
    const int nvec = (4 * TM * KPAD * 2) / 16;
    for (int i = tid; i < nvec; i += THREADS) p[i] = z;
  }
  __syncthreads();

  // Stage X tile: fp32 stream from HBM -> (hi,lo) f16 split in LDS.
  for (int i = tid; i < TM * (D_IN / 4); i += THREADS) {
    const int r  = i / (D_IN / 4);
    const int c4 = i - r * (D_IN / 4);
    long rg = row0 + r; if (rg >= natoms) rg = natoms - 1;
    const v4f v = *(const v4f*)(X + rg * D_IN + c4 * 4);
    _Float16* ph = Xhi + r * KPAD + c4 * 4;
    _Float16* pl = Xlo + r * KPAD + c4 * 4;
#pragma unroll
    for (int j = 0; j < 4; ++j) {
      float f = v[j];
      _Float16 h = (_Float16)f;
      ph[j] = h;
      pl[j] = (_Float16)(f - (float)h);
    }
  }
  __syncthreads();

  // Layer 1: H1 = relu(X @ W1 + b1)
  gemm_layer<false>(Xhi, Xlo, Wh, Wh + PER, b1, Hhi, Hlo, (float*)nullptr);
  __syncthreads();
  // Layer 2: H2 = relu(H1 @ W2 + b2), fp32 into LDS (X region is dead now)
  gemm_layer<true>(Hhi, Hlo, Wh + 2 * PER, Wh + 3 * PER, b2,
                   (_Float16*)nullptr, (_Float16*)nullptr, Hf);
  __syncthreads();

  // Layer 3: out[m] = H2[m,:] . W3 + b3   (600 FLOP/atom -> plain VALU fp32)
  if (tid < TM) {
    const v4f* hrow = (const v4f*)(Hf + tid * NPAD);
    const v4f* w3v  = (const v4f*)W3;
    float s = b3[0];
#pragma unroll 5
    for (int q = 0; q < D_HID / 4; ++q) {
      const v4f h = hrow[q], w = w3v[q];
      s += h[0] * w[0] + h[1] * w[1] + h[2] * w[2] + h[3] * w[3];
    }
    if (row0 + tid < natoms) out[row0 + tid] = s;
  }
}

// ---------------------------------------------------------------------------
extern "C" void kernel_launch(void* const* d_in, const int* in_sizes, int n_in,
                              void* d_out, int out_size, void* d_ws, size_t ws_size,
                              hipStream_t stream) {
  const float* x  = (const float*)d_in[0];
  const float* W1 = (const float*)d_in[1];
  const float* b1 = (const float*)d_in[2];
  const float* W2 = (const float*)d_in[3];
  const float* b2 = (const float*)d_in[4];
  const float* W3 = (const float*)d_in[5];
  const float* b3 = (const float*)d_in[6];
  float* out = (float*)d_out;

  const int natoms = in_sizes[0] / D_IN;            // 524288
  if (ws_size < (size_t)4 * PER * sizeof(_Float16)) return;  // ~778 KB needed

  _Float16* wh = (_Float16*)d_ws;
  prep_weights_f16split<<<(2 * PER + 255) / 256, 256, 0, stream>>>(W1, W2, wh);

  const int nblocks = (natoms + TM - 1) / TM;       // 8192
  const size_t smem = (size_t)4 * TM * KPAD * sizeof(_Float16);  // 163840 B
  mlp3_fused_wmma<<<nblocks, THREADS, smem, stream>>>(x, wh, b1, b2, W3, b3,
                                                      out, natoms);
}
